// AttentionalSplatting_15539191677462
// MI455X (gfx1250) — matrix operand order repaired and checked by reading, not compiled, not run on hardware
//
#include <hip/hip_runtime.h>

// ---------------------------------------------------------------------------
// AttentionalSplatting on MI455X (gfx1250, wave32, WMMA + TDM).
// T=16, M=256, HW=1024, D_MODEL=D_K=512, H=8, HEAD=64, SIGMA=0.5.
// All GEMM-like math via v_wmma_f32_16x16x32_bf16 (fp32 accum).
// K head-tiles staged to LDS via tensor_load_to_lds (TDM, TENSORcnt).
// ---------------------------------------------------------------------------

typedef __attribute__((ext_vector_type(16))) __bf16 v16bf;
typedef __attribute__((ext_vector_type(8)))  __bf16 v8bf;
typedef __attribute__((ext_vector_type(8)))  float  v8f;
typedef unsigned int u32x4 __attribute__((ext_vector_type(4)));
typedef int          i32x4 __attribute__((ext_vector_type(4)));
typedef int          i32x8 __attribute__((ext_vector_type(8)));

#define TT   16
#define MM   256
#define HWN  1024
#define DM   512
#define NH   8
#define HD   64

// ---- workspace layout (bytes), with aliasing of dead buffers ----
static constexpr size_t OFF_XQ = 0;          // 16 MB bf16 fpe     -> later Qn (bf16, same size)
static constexpr size_t OFF_XK = 16777216;   //  4 MB bf16 tpe     -> later Kn
static constexpr size_t OFF_XV = 20971520;   //  4 MB bf16 tokens
static constexpr size_t OFF_VT = 25165824;   //  4 MB bf16 V, transposed [T][H][dv][m]
static constexpr size_t OFF_WQ = 29360128;   // 512 KB bf16
static constexpr size_t OFF_WK = 29884416;
static constexpr size_t OFF_WV = 30408704;
static constexpr size_t OFF_WO = 30932992;
static constexpr size_t OFF_QF = 31457280;   // 32 MB f32 Q        -> later upd (bf16, 16 MB)
static constexpr size_t OFF_KF = 65011712;   //  8 MB f32 K
// total ~70 MB

// ---------------------------------------------------------------------------
__global__ __launch_bounds__(256) void cvt_f32_to_bf16(const float* __restrict__ s,
                                                       __bf16* __restrict__ d, int n) {
  int i = blockIdx.x * 256 + threadIdx.x;
  if (i < n) d[i] = (__bf16)s[i];
}

// ---------------------------------------------------------------------------
// TDM: DMA a 2D bf16 tile [tile_d1 rows x tile_d0 cols], row stride `stride0`
// elements, from global `gptr` into LDS at byte offset `lds_addr`.
// Descriptor packing per CDNA5 ISA 8.3/8.4 (count=1, type=2, data_size=2B).
// ---------------------------------------------------------------------------
__device__ __forceinline__ void tdm_load_2d_bf16(unsigned lds_addr, const void* gptr,
                                                 unsigned tensor_d0, unsigned tensor_d1,
                                                 unsigned tile_d0, unsigned tile_d1,
                                                 unsigned stride0) {
  unsigned long long ga = (unsigned long long)(size_t)gptr;
  u32x4 g0;
  g0[0] = 1u;                                              // count=1 (valid user D#)
  g0[1] = lds_addr;                                        // lds_addr[31:0]
  g0[2] = (unsigned)(ga & 0xFFFFFFFFu);                    // global_addr[31:0]
  g0[3] = (unsigned)((ga >> 32) & 0x01FFFFFFu)             // global_addr[56:32]
        | 0x80000000u;                                     // type=2 ("image")
  i32x8 g1;
  g1[0] = 0x00010000;                                      // data_size=1 (2 bytes)
  g1[1] = (int)(tensor_d0 << 16);                          // tensor_dim0[15:0] @ bits 63:48
  g1[2] = (int)((tensor_d0 >> 16) | (tensor_d1 << 16));    // dim0[31:16] | dim1[15:0]
  g1[3] = (int)((tensor_d1 >> 16) | (tile_d0 << 16));      // dim1[31:16] | tile_dim0
  g1[4] = (int)tile_d1;                                    // tile_dim1 (tile_dim2 = 0)
  g1[5] = (int)stride0;                                    // tensor_dim0_stride[31:0]
  g1[6] = 0;                                               // stride0 hi / dim1_stride lo
  g1[7] = 0;                                               // dim1_stride hi
  i32x4 gz = {0, 0, 0, 0};
#if defined(__clang_major__) && (__clang_major__ >= 23)
  i32x8 gz8 = {0, 0, 0, 0, 0, 0, 0, 0};
  __builtin_amdgcn_tensor_load_to_lds(g0, g1, gz, gz, gz8, 0);
#else
  __builtin_amdgcn_tensor_load_to_lds(g0, g1, gz, gz, 0);
#endif
}

// ---------------------------------------------------------------------------
// Y = X[rows,512] @ W[512,512]^T.  One wave -> 16x64 tile of Y.
// mode: 0 = store f32 row-major; 1 = store bf16 row-major;
//       2 = store bf16 transposed per (t,head): VT[((t*8+h)*64+dv)*256+m]
// A layout: lane=row M%16; j -> K=k+(hi?16:0)+j.
// B layout: lane=col N%16; j -> K=k+(hi?16:0)+j.   C: VGPR r -> rows r / r+8.
// ---------------------------------------------------------------------------
__global__ __launch_bounds__(256) void gemm512_wmma(const __bf16* __restrict__ X,
                                                    const __bf16* __restrict__ W,
                                                    void* __restrict__ Yv,
                                                    int rows, int mode) {
  const int lane = threadIdx.x & 31;
  const int l16  = lane & 15;
  const int hi   = lane >> 4;
  const int wid  = blockIdx.x * 8 + (threadIdx.x >> 5);
  const int mt   = wid >> 3;
  const int ng   = wid & 7;
  if (mt * 16 >= rows) return;                // wave-uniform

  const __bf16* xrow = X + (size_t)(mt * 16 + l16) * DM;
  v8f acc[4] = {};
  for (int k = 0; k < DM; k += 32) {
    v16bf a  = *(const v16bf*)(xrow + k + hi * 16);
#pragma unroll
    for (int nt = 0; nt < 4; ++nt) {
      const int ncol = ng * 64 + nt * 16 + l16;
      v16bf b = *(const v16bf*)(W + (size_t)ncol * DM + k + hi * 16);
      acc[nt] = __builtin_amdgcn_wmma_f32_16x16x32_bf16(false, a, false, b,
                                                        (short)0, acc[nt], false, false);
    }
    if (k + 32 < DM) __builtin_prefetch(xrow + k + 32, 0, 1);  // global_prefetch_b8
  }
  if (mode == 0) {
    float* Y = (float*)Yv;
#pragma unroll
    for (int nt = 0; nt < 4; ++nt)
#pragma unroll
      for (int r = 0; r < 8; ++r)
        Y[(size_t)(mt * 16 + hi * 8 + r) * DM + ng * 64 + nt * 16 + l16] = acc[nt][r];
  } else if (mode == 1) {
    __bf16* Y = (__bf16*)Yv;
#pragma unroll
    for (int nt = 0; nt < 4; ++nt)
#pragma unroll
      for (int r = 0; r < 8; ++r)
        Y[(size_t)(mt * 16 + hi * 8 + r) * DM + ng * 64 + nt * 16 + l16] = (__bf16)acc[nt][r];
  } else {
    // V-transpose store: rows are (t*256+m), cols are (h*64+dv); h == ng here.
    __bf16* Y = (__bf16*)Yv;
#pragma unroll
    for (int nt = 0; nt < 4; ++nt)
#pragma unroll
      for (int r = 0; r < 8; ++r) {
        const int row = mt * 16 + hi * 8 + r;
        const int t   = row >> 8;
        const int m   = row & (MM - 1);
        Y[(((size_t)t * NH + ng) * HD + nt * 16 + l16) * MM + m] = (__bf16)acc[nt][r];
      }
  }
}

// ---------------------------------------------------------------------------
// Row LayerNorm (no bias) over 512, one wave32 per row, bf16 output.
// ---------------------------------------------------------------------------
__global__ __launch_bounds__(256) void ln512_bf16(const float* __restrict__ Y,
                                                  const float* __restrict__ gamma,
                                                  __bf16* __restrict__ out, int rows) {
  const int lane = threadIdx.x & 31;
  const int row  = blockIdx.x * 8 + (threadIdx.x >> 5);
  if (row >= rows) return;
  const float* y = Y + (size_t)row * DM;
  float v[16];
  float s = 0.f;
#pragma unroll
  for (int j = 0; j < 16; ++j) { v[j] = y[lane + j * 32]; s += v[j]; }
#pragma unroll
  for (int off = 16; off >= 1; off >>= 1) s += __shfl_xor(s, off, 32);
  const float mu = s * (1.0f / 512.0f);
  float var = 0.f;
#pragma unroll
  for (int j = 0; j < 16; ++j) { float d = v[j] - mu; var += d * d; }
#pragma unroll
  for (int off = 16; off >= 1; off >>= 1) var += __shfl_xor(var, off, 32);
  const float rs = rsqrtf(var * (1.0f / 512.0f) + 1e-6f);
#pragma unroll
  for (int j = 0; j < 16; ++j)
    out[(size_t)row * DM + lane + j * 32] = (__bf16)((v[j] - mu) * rs * gamma[lane + j * 32]);
}

// ---------------------------------------------------------------------------
// Attention: one wave per (t, head, 16-query tile).
//   K head-slice [256 x 64] bf16 DMA'd to LDS by the TDM (tensor_load_to_lds).
//   scores[16,256] = Qh Kh^T / 8 (32 WMMAs, B from LDS) -> LDS
//   spatial bias -2*d2 (sigma=0.5) + stable softmax over 256 keys
//   upd[16,64] = P @ Vh (32 WMMAs, B = contiguous loads from pre-transposed VT)
// ---------------------------------------------------------------------------
__global__ __launch_bounds__(32) void attn_wmma(const __bf16* __restrict__ Qn,
                                                const __bf16* __restrict__ Kn,
                                                const __bf16* __restrict__ VT,
                                                const float*  __restrict__ tracks,
                                                const float*  __restrict__ fpos,
                                                __bf16* __restrict__ upd) {
  __shared__ __align__(32) __bf16 s_k[MM * HD];   // 32 KB K tile (TDM destination)
  __shared__ __align__(32) float  s_s[16 * MM];   // 16 KB scores / exp
  __shared__ __align__(32) __bf16 s_p[16 * MM];   //  8 KB probabilities
  const int qt = blockIdx.x, h = blockIdx.y, t = blockIdx.z;
  const int lane = threadIdx.x;
  const int l16 = lane & 15, hi = lane >> 4;

  // ---- TDM: K[t, :, h*64 : h*64+64] -> LDS (row stride 512 elements) ----
  tdm_load_2d_bf16((unsigned)(size_t)(void*)s_k,
                   Kn + (size_t)t * MM * DM + h * HD,
                   HD, MM, HD, MM, DM);

  // ---- Q tile (A operand, contiguous k = 16*hi + j) while the DMA flies ----
  const __bf16* qrow = Qn + ((size_t)t * HWN + qt * 16 + l16) * DM + h * HD;
  v16bf a0 = *(const v16bf*)(qrow + 0  + hi * 16);
  v16bf a1 = *(const v16bf*)(qrow + 32 + hi * 16);

  __builtin_amdgcn_s_wait_tensorcnt(0);           // K tile resident in LDS

  // ---- QK^T scores, B operands read from LDS ----
#pragma unroll
  for (int nt = 0; nt < 16; ++nt) {
    const __bf16* kr = s_k + (size_t)(nt * 16 + l16) * HD;
    v16bf b0 = *(const v16bf*)(kr + hi * 16);
    v16bf b1 = *(const v16bf*)(kr + 32 + hi * 16);
    v8f c = {};
    c = __builtin_amdgcn_wmma_f32_16x16x32_bf16(false, a0, false, b0, (short)0, c, false, false);
    c = __builtin_amdgcn_wmma_f32_16x16x32_bf16(false, a1, false, b1, (short)0, c, false, false);
#pragma unroll
    for (int r = 0; r < 8; ++r)
      s_s[(hi * 8 + r) * MM + nt * 16 + l16] = c[r] * 0.125f;   // 1/sqrt(64)
  }
  __syncthreads();

  // ---- bias + softmax: row = l16, this lane covers cols [hi*128, hi*128+128) ----
  const float fx = fpos[(qt * 16 + l16) * 2 + 0];
  const float fy = fpos[(qt * 16 + l16) * 2 + 1];
  const float* trk = tracks + (size_t)t * MM * 2;
  float mx = -3.0e38f;
  for (int c = 0; c < 128; ++c) {
    const int m = hi * 128 + c;
    const float dx = fx - trk[m * 2 + 0];
    const float dy = fy - trk[m * 2 + 1];
    const float sv = s_s[l16 * MM + m] - 2.0f * (dx * dx + dy * dy); // -d2/(2*sigma^2)
    s_s[l16 * MM + m] = sv;
    mx = fmaxf(mx, sv);
  }
  mx = fmaxf(mx, __shfl_xor(mx, 16, 32));
  float sum = 0.f;
  for (int c = 0; c < 128; ++c) {
    const int m = hi * 128 + c;
    const float e = __expf(s_s[l16 * MM + m] - mx);
    s_s[l16 * MM + m] = e;
    sum += e;
  }
  sum += __shfl_xor(sum, 16, 32);
  const float inv = 1.0f / sum;
  for (int c = 0; c < 128; ++c) {
    const int m = hi * 128 + c;
    s_p[l16 * MM + m] = (__bf16)(s_s[l16 * MM + m] * inv);
  }
  __syncthreads();

  // ---- P @ Vh : B operands are contiguous rows of pre-transposed VT ----
  v8f o[4] = {};
  const __bf16* vth = VT + ((size_t)t * NH + h) * HD * MM;   // [dv][m], m contiguous
#pragma unroll
  for (int kc = 0; kc < 8; ++kc) {
    const __bf16* pr = &s_p[l16 * MM + kc * 32 + hi * 16];
    v16bf a = *(const v16bf*)pr;
#pragma unroll
    for (int nt = 0; nt < 4; ++nt) {
      v16bf b = *(const v16bf*)(vth + (size_t)(nt * 16 + l16) * MM + kc * 32 + hi * 16);
      o[nt] = __builtin_amdgcn_wmma_f32_16x16x32_bf16(false, a, false, b,
                                                      (short)0, o[nt], false, false);
    }
  }
  __bf16* up = upd + ((size_t)t * HWN + qt * 16 + hi * 8) * DM + h * HD + l16;
#pragma unroll
  for (int nt = 0; nt < 4; ++nt)
#pragma unroll
    for (int r = 0; r < 8; ++r)
      up[(size_t)r * DM + nt * 16] = (__bf16)o[nt][r];
}

// ---------------------------------------------------------------------------
extern "C" void kernel_launch(void* const* d_in, const int* in_sizes, int n_in,
                              void* d_out, int out_size, void* d_ws, size_t ws_size,
                              hipStream_t stream) {
  (void)in_sizes; (void)n_in; (void)out_size; (void)ws_size;
  const float* tok  = (const float*)d_in[0];   // updated_track_tokens [T,M,512]
  const float* trk  = (const float*)d_in[1];   // tracks               [T,M,2]
  const float* fps  = (const float*)d_in[2];   // feature_positions    [HW,2]
  const float* fpe  = (const float*)d_in[3];   // feature_pos_emb      [T,HW,512]
  const float* tpe  = (const float*)d_in[4];   // track_pos_emb        [T,M,512]
  const float* Wq   = (const float*)d_in[5];
  const float* Wk   = (const float*)d_in[6];
  const float* Wv   = (const float*)d_in[7];
  const float* Wo   = (const float*)d_in[8];
  const float* qg   = (const float*)d_in[9];
  const float* kg   = (const float*)d_in[10];

  char* ws = (char*)d_ws;
  __bf16* xq  = (__bf16*)(ws + OFF_XQ);     // also Qn later
  __bf16* xk  = (__bf16*)(ws + OFF_XK);     // also Kn later
  __bf16* xv  = (__bf16*)(ws + OFF_XV);
  __bf16* vt  = (__bf16*)(ws + OFF_VT);
  __bf16* wqb = (__bf16*)(ws + OFF_WQ);
  __bf16* wkb = (__bf16*)(ws + OFF_WK);
  __bf16* wvb = (__bf16*)(ws + OFF_WV);
  __bf16* wob = (__bf16*)(ws + OFF_WO);
  float*  qf  = (float* )(ws + OFF_QF);
  float*  kf  = (float* )(ws + OFF_KF);
  __bf16* qn  = xq;                          // alias: fpe-bf16 dead after Q gemm
  __bf16* kn  = xk;                          // alias: tpe-bf16 dead after K gemm
  __bf16* ub  = (__bf16*)(ws + OFF_QF);      // alias: Q-f32 dead after layernorm

  const int nQ = TT * HWN * DM, nK = TT * MM * DM, nW = DM * DM;
  cvt_f32_to_bf16<<<(nQ + 255) / 256, 256, 0, stream>>>(fpe, xq, nQ);
  cvt_f32_to_bf16<<<(nK + 255) / 256, 256, 0, stream>>>(tpe, xk, nK);
  cvt_f32_to_bf16<<<(nK + 255) / 256, 256, 0, stream>>>(tok, xv, nK);
  cvt_f32_to_bf16<<<(nW + 255) / 256, 256, 0, stream>>>(Wq, wqb, nW);
  cvt_f32_to_bf16<<<(nW + 255) / 256, 256, 0, stream>>>(Wk, wkb, nW);
  cvt_f32_to_bf16<<<(nW + 255) / 256, 256, 0, stream>>>(Wv, wvb, nW);
  cvt_f32_to_bf16<<<(nW + 255) / 256, 256, 0, stream>>>(Wo, wob, nW);

  const int rowsQ = TT * HWN, rowsK = TT * MM;
  gemm512_wmma<<<rowsQ / 16, 256, 0, stream>>>(xq, wqb, qf, rowsQ, 0); // Q f32
  gemm512_wmma<<<rowsK / 16, 256, 0, stream>>>(xk, wkb, kf, rowsK, 0); // K f32
  gemm512_wmma<<<rowsK / 16, 256, 0, stream>>>(xv, wvb, vt, rowsK, 2); // V bf16 transposed

  ln512_bf16<<<rowsQ / 8, 256, 0, stream>>>(qf, qg, qn, rowsQ);
  ln512_bf16<<<rowsK / 8, 256, 0, stream>>>(kf, kg, kn, rowsK);

  attn_wmma<<<dim3(HWN / 16, NH, TT), 32, 0, stream>>>(qn, kn, vt, trk, fps, ub);

  gemm512_wmma<<<rowsQ / 16, 256, 0, stream>>>(ub, wob, d_out, rowsQ, 0); // fp32 out
}
